// GCNNET_gpool_32083405701287
// MI455X (gfx1250) — 16-bit and 8-bit matrix operand order repaired and checked by reading, compiled, not run on hardware
//
#include <hip/hip_runtime.h>
#include <hip/hip_bf16.h>


#define USE_ASYNC_LDS 1

typedef int      v8i  __attribute__((ext_vector_type(8)));
typedef float    v8f  __attribute__((ext_vector_type(8)));
typedef _Float16 v16h __attribute__((ext_vector_type(16)));

struct alignas(8)  U2 { unsigned x, y; };
struct alignas(16) U4 { unsigned x, y, z, w; };

union IFrag { v8i v; int i[8]; unsigned u[8]; U2 d[4]; U4 q[2]; };
union FFrag { v8f v; float f[8]; };
union HFrag { v16h v; _Float16 h[16]; unsigned u[8]; U4 q[2]; };

#define NB 256
#define ROI 500
#define NP0 512

__global__ void k_start_gcn(const float* __restrict__ g,
                            const float* __restrict__ W0,
                            const float* __restrict__ b0,
                            float* __restrict__ F,
                            unsigned char* __restrict__ ub)
{
    int s = blockIdx.x;
    int r = blockIdx.y;
    const float* row = g + ((size_t)s * ROI + r) * ROI;
    unsigned char* ubrow = ub + ((size_t)s * NP0 + r) * NP0;
    __shared__ float red[256];
    float acc = 0.f;
    for (int c = threadIdx.x; c < NP0; c += 256) {
        float v = 0.f;
        if (r < ROI && c < ROI) v = row[c];
        ubrow[c] = (v != 0.f) ? (unsigned char)1 : (unsigned char)0;
        if (c < ROI) acc += v * W0[c];
    }
    red[threadIdx.x] = acc;
    __syncthreads();
    for (int st = 128; st > 0; st >>= 1) {
        if (threadIdx.x < st) red[threadIdx.x] += red[threadIdx.x + st];
        __syncthreads();
    }
    if (threadIdx.x == 0 && r < ROI) {
        float h = red[0] + b0[0];
        F[(size_t)s * 1500 + r] = h > 0.f ? h : 0.f;
    }
}

__global__ void k_topk(const float* __restrict__ F, int foff, int n, int nk,
                       const float* __restrict__ Wp, const float* __restrict__ bp,
                       int lvl,
                       int* __restrict__ idx_out,
                       float* __restrict__ newh_out)
{
    int s = blockIdx.x;
    __shared__ float sc[512];
    __shared__ float hv[512];
    float wp = Wp[lvl], bpv = bp[lvl];
    int t = threadIdx.x;
    if (t < n) {
        float h = F[(size_t)s * 1500 + foff + t];
        hv[t] = h;
        float z = h * wp + bpv;
        sc[t] = 1.f / (1.f + __expf(-z));
    }
    __syncthreads();
    if (t < n) {
        float sj = sc[t];
        int rank = 0;
        for (int l = 0; l < n; ++l) {
            float sl = sc[l];
            rank += ((sl > sj) || (sl == sj && l < t)) ? 1 : 0;
        }
        if (rank < nk) {
            idx_out[(size_t)s * 400 + rank] = t;
            newh_out[(size_t)s * 400 + rank] = hv[t] * sj;
        }
    }
}

__global__ void k_bt_gather(const unsigned char* __restrict__ ub_in,
                            int np_in, int n_in,
                            const int* __restrict__ idx, int nk, int np_out,
                            unsigned char* __restrict__ BT)
{
    int s = blockIdx.y;
    int j = blockIdx.x * 32 + (threadIdx.x >> 3);
    int lane8 = threadIdx.x & 7;
    int col = (j < nk) ? idx[(size_t)s * 400 + j] : n_in;
    const unsigned char* G = ub_in + (size_t)s * np_in * np_in;
    unsigned char* Bj = BT + ((size_t)s * np_out + j) * np_in;
    for (int k0 = lane8 * 4; k0 < np_in; k0 += 32) {
        unsigned b0 = G[(size_t)(k0 + 0) * np_in + col];
        unsigned b1 = G[(size_t)(k0 + 1) * np_in + col];
        unsigned b2 = G[(size_t)(k0 + 2) * np_in + col];
        unsigned b3 = G[(size_t)(k0 + 3) * np_in + col];
        *(unsigned*)(Bj + k0) = b0 | (b1 << 8) | (b2 << 16) | (b3 << 24);
    }
}

__device__ __forceinline__ void ld_iu8(const unsigned char* Arow,
                                       const unsigned char* Brow,
                                       int kb, int hi, IFrag& a, IFrag& b)
{
    a.q[0] = *(const U4*)(Arow + kb + hi * 16);
    a.q[1] = *(const U4*)(Arow + kb + 32 + hi * 16);
    b.q[0] = *(const U4*)(Brow + kb + hi * 16);
    b.q[1] = *(const U4*)(Brow + kb + 32 + hi * 16);
}

__global__ void k_twohop_wmma(const unsigned char* __restrict__ ub_in,
                              int np_in, int n_in,
                              const int* __restrict__ idx, int nk, int np_out,
                              const unsigned char* __restrict__ BT,
                              unsigned char* __restrict__ ub_out)
{
    int tilesM = np_out >> 4;
    int tiles_per_sample = tilesM * tilesM;
    int wave = threadIdx.x >> 5;
    int lane = threadIdx.x & 31;
    int tl = blockIdx.x * 4 + wave;
    int s = tl / tiles_per_sample;
    if (s >= NB) return;
    int rem = tl % tiles_per_sample;
    int tm = rem / tilesM, tn = rem % tilesM;
    int hi  = lane >> 4;
    int l15 = lane & 15;
    int m    = tm * 16 + l15;
    int ncol = tn * 16 + l15;
    const int* sidx = idx + (size_t)s * 400;
    int rowA = (m < nk) ? sidx[m] : n_in;
    const unsigned char* Arow = ub_in + (size_t)s * np_in * np_in
                                      + (size_t)rowA * np_in;
    const unsigned char* Brow = BT + ((size_t)s * np_out + ncol) * np_in;

    IFrag c0, c1;
    c0.v = (v8i){0,0,0,0,0,0,0,0};
    c1.v = (v8i){0,0,0,0,0,0,0,0};
    for (int kb = 0; kb < np_in; kb += 128) {
        IFrag a, b;
        ld_iu8(Arow, Brow, kb, hi, a, b);
        c0.v = __builtin_amdgcn_wmma_i32_16x16x64_iu8(false, a.v, false, b.v,
                                                      c0.v, false, false);
        ld_iu8(Arow, Brow, kb + 64, hi, a, b);
        c1.v = __builtin_amdgcn_wmma_i32_16x16x64_iu8(false, a.v, false, b.v,
                                                      c1.v, false, false);
    }
    unsigned char* Orow = ub_out + (size_t)s * np_out * np_out;
    bool nvalid = (ncol < nk);
#pragma unroll
    for (int v = 0; v < 8; ++v) {
        int mo = tm * 16 + v + (hi << 3);
        int sum = c0.i[v] + c1.i[v];
        Orow[(size_t)mo * np_out + ncol] =
            (mo < nk && nvalid && sum > 0) ? (unsigned char)1 : (unsigned char)0;
    }
}

__global__ void k_gcn_down(const unsigned char* __restrict__ ub, int nk, int np,
                           const float* __restrict__ newh,
                           const float* __restrict__ Wg,
                           const float* __restrict__ bg, int lvl,
                           float* __restrict__ F, int foff_out)
{
    int s = blockIdx.x;
    __shared__ float nhsh[400];
    __shared__ float tsh[400];
    const unsigned char* G = ub + (size_t)s * np * np;
    const float* nh = newh + (size_t)s * 400;
    int t = threadIdx.x;
#if USE_ASYNC_LDS
    for (int j = t; j < nk; j += blockDim.x) {
        unsigned lds_off = (unsigned)(unsigned long long)&nhsh[j];
        unsigned long long ga = (unsigned long long)(const void*)(nh + j);
        asm volatile("global_load_async_to_lds_b32 %0, %1, off"
                     :: "v"(lds_off), "v"(ga) : "memory");
    }
    asm volatile("s_wait_asynccnt 0" ::: "memory");
#else
    for (int j = t; j < nk; j += blockDim.x) nhsh[j] = nh[j];
#endif
    __syncthreads();
    for (int j = t; j < nk; j += blockDim.x) {
        const unsigned char* rowj = G + (size_t)j * np;
        int d = 0;
        for (int k2 = 0; k2 < nk; ++k2) d += rowj[k2];
        tsh[j] = (d > 0) ? (nhsh[j] / (float)d) : 0.f;
    }
    __syncthreads();
    float wg = Wg[lvl], bgv = bg[lvl];
    for (int r = t; r < nk; r += blockDim.x) {
        const unsigned char* rowr = G + (size_t)r * np;
        float acc = 0.f;
        for (int j = 0; j < nk; ++j) acc += rowr[j] ? tsh[j] : 0.f;
        float h = acc * wg + bgv;
        F[(size_t)s * 1500 + foff_out + r] = h > 0.f ? h : 0.f;
    }
}

__global__ void k_zero_u16(unsigned short* __restrict__ p, long nelem)
{
    long i = (long)blockIdx.x * blockDim.x + threadIdx.x;
    if (i < nelem) p[i] = 0;
}

__global__ void k_wpad_t(const float* __restrict__ W, int K, int N,
                         _Float16* __restrict__ WhT, int Kp, int Np)
{
    int k = blockIdx.x * 256 + threadIdx.x;
    int n = blockIdx.y;
    if (k < Kp && n < Np) {
        float v = (k < K && n < N) ? W[(size_t)k * N + n] : 0.f;
        WhT[(size_t)n * Kp + k] = (_Float16)v;
    }
}

__global__ void k_bn_relu_f16(const float* __restrict__ X, int Fdim, int Fp,
                              const float* __restrict__ gamma,
                              const float* __restrict__ beta,
                              _Float16* __restrict__ Xh)
{
    int f = blockIdx.x;
    int t = threadIdx.x;
    __shared__ float s1[256], s2[256];
    __shared__ float mu, rs;
    float x = X[(size_t)t * Fdim + f];
    s1[t] = x; s2[t] = x * x;
    __syncthreads();
    for (int st = 128; st > 0; st >>= 1) {
        if (t < st) { s1[t] += s1[t + st]; s2[t] += s2[t + st]; }
        __syncthreads();
    }
    if (t == 0) {
        float m = s1[0] * (1.f / 256.f);
        float var = s2[0] * (1.f / 256.f) - m * m;
        mu = m; rs = rsqrtf(var + 1e-5f);
    }
    __syncthreads();
    float y = (x - mu) * rs * gamma[f] + beta[f];
    y = y > 0.f ? y : 0.f;
    Xh[(size_t)t * Fp + f] = (_Float16)y;
}

__global__ void k_gemm_f16(const _Float16* __restrict__ A,
                           const _Float16* __restrict__ Bt,
                           const float* __restrict__ bias, int biasN,
                           int Kp, int N, float* __restrict__ C)
{
    int tilesN = N >> 4;
    int wave = threadIdx.x >> 5;
    int lane = threadIdx.x & 31;
    int tl = blockIdx.x * 4 + wave;
    if (tl >= 16 * tilesN) return;
    int tm = tl / tilesN, tn = tl % tilesN;
    int hi  = lane >> 4;
    int l15 = lane & 15;
    int m = tm * 16 + l15;
    int n = tn * 16 + l15;
    const _Float16* Arow = A + (size_t)m * Kp;
    const _Float16* Brow = Bt + (size_t)n * Kp;

    FFrag c0, c1;
    c0.v = (v8f){0.f,0.f,0.f,0.f,0.f,0.f,0.f,0.f};
    c1.v = (v8f){0.f,0.f,0.f,0.f,0.f,0.f,0.f,0.f};
    for (int kb = 0; kb < Kp; kb += 64) {
        if (kb + 128 < Kp)
            __builtin_prefetch(Arow + kb + 128, 0, 1);
        HFrag a, b;
        a.q[0] = *(const U4*)(Arow + kb + hi * 8);
        a.q[1] = *(const U4*)(Arow + kb + 16 + hi * 8);
        b.q[0] = *(const U4*)(Brow + kb + hi * 8);
        b.q[1] = *(const U4*)(Brow + kb + 16 + hi * 8);
        c0.v = __builtin_amdgcn_wmma_f32_16x16x32_f16(false, a.v, false, b.v,
                                                      (short)0, c0.v, false, false);
        a.q[0] = *(const U4*)(Arow + kb + 32 + hi * 8);
        a.q[1] = *(const U4*)(Arow + kb + 48 + hi * 8);
        b.q[0] = *(const U4*)(Brow + kb + 32 + hi * 8);
        b.q[1] = *(const U4*)(Brow + kb + 48 + hi * 8);
        c1.v = __builtin_amdgcn_wmma_f32_16x16x32_f16(false, a.v, false, b.v,
                                                      (short)0, c1.v, false, false);
    }
    float bv = (n < biasN) ? bias[n] : 0.f;
#pragma unroll
    for (int v = 0; v < 8; ++v) {
        int mo = tm * 16 + v + (hi << 3);
        C[(size_t)mo * N + n] = c0.f[v] + c1.f[v] + bv;
    }
}

__global__ void k_softmax2(const float* __restrict__ H, float* __restrict__ out)
{
    int s = blockIdx.x * blockDim.x + threadIdx.x;
    if (s < NB) {
        float a = H[(size_t)s * 16 + 0];
        float b = H[(size_t)s * 16 + 1];
        float mx = fmaxf(a, b);
        float ea = __expf(a - mx), eb = __expf(b - mx);
        float inv = 1.f / (ea + eb);
        out[s * 2 + 0] = ea * inv;
        out[s * 2 + 1] = eb * inv;
    }
}

extern "C" void kernel_launch(void* const* d_in, const int* in_sizes, int n_in,
                              void* d_out, int out_size, void* d_ws, size_t ws_size,
                              hipStream_t stream)
{
    (void)in_sizes; (void)n_in; (void)out_size; (void)ws_size;

    const float* g   = (const float*)d_in[0];
    const float* W0  = (const float*)d_in[1];
    const float* b0  = (const float*)d_in[2];
    const float* Wp  = (const float*)d_in[3];
    const float* bp  = (const float*)d_in[4];
    const float* Wg  = (const float*)d_in[5];
    const float* bg  = (const float*)d_in[6];
    const float* gamma1 = (const float*)d_in[7];
    const float* beta1  = (const float*)d_in[8];
    const float* W1  = (const float*)d_in[9];
    const float* b1  = (const float*)d_in[10];
    const float* gamma2 = (const float*)d_in[11];
    const float* beta2  = (const float*)d_in[12];
    const float* W2  = (const float*)d_in[13];
    const float* b2  = (const float*)d_in[14];
    const float* gamma3 = (const float*)d_in[15];
    const float* beta3  = (const float*)d_in[16];
    const float* W3  = (const float*)d_in[17];
    const float* b3  = (const float*)d_in[18];
    const float* gamma4 = (const float*)d_in[19];
    const float* beta4  = (const float*)d_in[20];
    const float* W4  = (const float*)d_in[21];
    const float* b4  = (const float*)d_in[22];
    float* out = (float*)d_out;

    char* ws = (char*)d_ws;
    size_t off = 0;
    auto carve = [&](size_t bytes) { size_t o = off; off = (off + bytes + 255) & ~(size_t)255; return ws + o; };
    unsigned char* ubA  = (unsigned char*)carve((size_t)NB * NP0 * NP0);
    unsigned char* ubB  = (unsigned char*)carve((size_t)NB * NP0 * NP0);
    unsigned char* BT   = (unsigned char*)carve((size_t)NB * NP0 * NP0);
    float*         F    = (float*)carve((size_t)NB * 1500 * sizeof(float));
    int*           idxb = (int*)carve((size_t)NB * 400 * sizeof(int));
    float*         newh = (float*)carve((size_t)NB * 400 * sizeof(float));
    _Float16*      Xh   = (_Float16*)carve((size_t)NB * 1536 * sizeof(_Float16));
    float*         Ha   = (float*)carve((size_t)NB * 512 * sizeof(float));
    float*         Hb   = (float*)carve((size_t)NB * 512 * sizeof(float));
    _Float16*      W1h  = (_Float16*)carve((size_t)512 * 1536 * sizeof(_Float16));
    _Float16*      W2h  = (_Float16*)carve((size_t)256 * 512 * sizeof(_Float16));
    _Float16*      W3h  = (_Float16*)carve((size_t)128 * 256 * sizeof(_Float16));
    _Float16*      W4h  = (_Float16*)carve((size_t)16 * 128 * sizeof(_Float16));

    k_start_gcn<<<dim3(NB, NP0), 256, 0, stream>>>(g, W0, b0, F, ubA);

    const int nlv[4]    = {500, 400, 300, 200};
    const int nplv[4]   = {512, 512, 384, 256};
    const int nklv[4]   = {400, 300, 200, 100};
    const int npolv[4]  = {512, 384, 256, 128};
    const int fin[4]    = {0, 500, 900, 1200};
    const int fout[4]   = {500, 900, 1200, 1400};
    for (int lvl = 0; lvl < 4; ++lvl) {
        int n = nlv[lvl], np = nplv[lvl], nk = nklv[lvl], npo = npolv[lvl];
        unsigned char* uin  = (lvl & 1) ? ubB : ubA;
        unsigned char* uout = (lvl & 1) ? ubA : ubB;
        k_topk<<<NB, 512, 0, stream>>>(F, fin[lvl], n, nk, Wp, bp, lvl, idxb, newh);
        k_bt_gather<<<dim3(npo / 32, NB), 256, 0, stream>>>(uin, np, n, idxb, nk, npo, BT);
        int tilesM = npo >> 4;
        int waves  = NB * tilesM * tilesM;
        k_twohop_wmma<<<(waves + 3) / 4, 128, 0, stream>>>(uin, np, n, idxb, nk, npo, BT, uout);
        k_gcn_down<<<NB, 256, 0, stream>>>(uout, nk, npo, newh, Wg, bg, lvl, F, fout[lvl]);
    }

    k_wpad_t<<<dim3(6, 512), 256, 0, stream>>>(W1, 1500, 512, W1h, 1536, 512);
    k_wpad_t<<<dim3(2, 256), 256, 0, stream>>>(W2, 512, 256, W2h, 512, 256);
    k_wpad_t<<<dim3(1, 128), 256, 0, stream>>>(W3, 256, 128, W3h, 256, 128);
    k_wpad_t<<<dim3(1, 16), 256, 0, stream>>>(W4, 128, 2, W4h, 128, 16);

    long xh_elems = (long)NB * 1536;
    k_zero_u16<<<(unsigned)((xh_elems + 255) / 256), 256, 0, stream>>>((unsigned short*)Xh, xh_elems);

    k_bn_relu_f16<<<1500, 256, 0, stream>>>(F, 1500, 1536, gamma1, beta1, Xh);
    { int tiles = 16 * (512 / 16); k_gemm_f16<<<(tiles + 3) / 4, 128, 0, stream>>>(Xh, W1h, b1, 512, 1536, 512, Ha); }
    k_bn_relu_f16<<<512, 256, 0, stream>>>(Ha, 512, 512, gamma2, beta2, Xh);
    { int tiles = 16 * (256 / 16); k_gemm_f16<<<(tiles + 3) / 4, 128, 0, stream>>>(Xh, W2h, b2, 256, 512, 256, Hb); }
    k_bn_relu_f16<<<256, 256, 0, stream>>>(Hb, 256, 256, gamma3, beta3, Xh);
    { int tiles = 16 * (128 / 16); k_gemm_f16<<<(tiles + 3) / 4, 128, 0, stream>>>(Xh, W3h, b3, 128, 256, 128, Ha); }
    k_bn_relu_f16<<<128, 256, 0, stream>>>(Ha, 128, 128, gamma4, beta4, Xh);
    { int tiles = 16 * 1; k_gemm_f16<<<(tiles + 3) / 4, 128, 0, stream>>>(Xh, W4h, b4, 2, 128, 16, Hb); }

    k_softmax2<<<1, 256, 0, stream>>>(Hb, out);
}
